// StreamingQwenMoE_72928544686527
// MI455X (gfx1250) — matrix operand order repaired and checked by reading, not compiled, not run on hardware
//
#include <hip/hip_runtime.h>

// ---------------------------------------------------------------------------
// MoE (Qwen-style) for gfx1250: bf16 WMMA 16x16x32 GEMMs, f32 accumulate.
// A-tiles staged via GLOBAL_LOAD_ASYNC_TO_LDS when available.
// ---------------------------------------------------------------------------

typedef __attribute__((ext_vector_type(16))) __bf16       v16bf;
typedef __attribute__((ext_vector_type(8)))  float        v8f;
typedef __attribute__((ext_vector_type(4)))  unsigned int u32x4;
typedef __attribute__((ext_vector_type(4)))  int          i32x4;
typedef __attribute__((ext_vector_type(4)))  float        f32x4;

#define T_     2048
#define D_     2048
#define DF_    768
#define DSH_   2048
#define E_     32
#define TOPK_  8
#define CAP_   1024
#define LPAD   40   // LDS row stride in halves (32 + 8 pad), 80B rows, 16B aligned

#if __has_builtin(__builtin_amdgcn_global_load_async_to_lds_b128)
#define ASYNC_A_TILE 1
typedef __attribute__((address_space(1))) i32x4* gptr_i4;   // global int4*
typedef __attribute__((address_space(3))) i32x4* lptr_i4;   // LDS int4*
#if __has_builtin(__builtin_amdgcn_s_wait_asynccnt)
#define ASYNC_WAIT() __builtin_amdgcn_s_wait_asynccnt(0)
#else
#define ASYNC_WAIT() asm volatile("s_wait_asynccnt 0" ::: "memory")
#endif
#else
#define ASYNC_A_TILE 0
#define ASYNC_WAIT()
#endif

union Frag16 { u32x4 q[2]; v16bf v; };
union H8     { unsigned short h[8]; u32x4 q; };

__device__ __forceinline__ unsigned short f2bf(float f) {
  unsigned int u = __float_as_uint(f);
  u += 0x7FFFu + ((u >> 16) & 1u);          // round-to-nearest-even
  return (unsigned short)(u >> 16);
}

__device__ __forceinline__ float fast_sigmoid(float x) {
#if __has_builtin(__builtin_amdgcn_rcpf)
  return __builtin_amdgcn_rcpf(1.f + __expf(-x));   // v_rcp_f32
#else
  return 1.f / (1.f + __expf(-x));
#endif
}

__device__ __forceinline__ v8f vzero8() {
  v8f z;
#pragma unroll
  for (int i = 0; i < 8; ++i) z[i] = 0.f;
  return z;
}

__device__ __forceinline__ v8f wmma_bf16(const Frag16& a, const Frag16& b, v8f c) {
  return __builtin_amdgcn_wmma_f32_16x16x32_bf16(false, a.v, false, b.v,
                                                 (short)0, c, false, false);
}

// --- cooperative tile loads (256 threads) ----------------------------------
// copy 32 bytes global(bf16) -> LDS for one (row,half) assignment
__device__ __forceinline__ void copyA32B(unsigned short* dst, const unsigned short* src) {
#if ASYNC_A_TILE
  __builtin_amdgcn_global_load_async_to_lds_b128(
      (gptr_i4)(unsigned short*)(src),     (lptr_i4)(dst),     0, 0);
  __builtin_amdgcn_global_load_async_to_lds_b128(
      (gptr_i4)(unsigned short*)(src + 8), (lptr_i4)(dst + 8), 0, 0);
#else
  u32x4 d0 = *(const u32x4*)(src);
  u32x4 d1 = *(const u32x4*)(src + 8);
  *(u32x4*)(dst)     = d0;
  *(u32x4*)(dst + 8) = d1;
#endif
}

// A-tile: 128 rows x 32 halves (bf16 source). thread -> (row=tid>>1, half=tid&1)
__device__ __forceinline__ void loadAtileDirect(unsigned short* lA,
                                                const unsigned short* A, int lda,
                                                int mBase, int kb, int tid) {
  const int r = tid >> 1, hh = tid & 1;
  const unsigned short* src = A + (size_t)(mBase + r) * lda + kb + hh * 16;
  copyA32B(lA + r * LPAD + hh * 16, src);
}

__device__ __forceinline__ void loadAtileGather(unsigned short* lA,
                                                const unsigned short* xb,
                                                const int* slots, int count,
                                                int mBase, int kb, int tid) {
  const int r = tid >> 1, hh = tid & 1;
  const int slot = mBase + r;
  const int b = (slot < count) ? slots[slot] : 0;   // flat t*K+k index
  const int tok = b >> 3;                           // TOPK_==8
  const unsigned short* src = xb + (size_t)tok * D_ + kb + hh * 16;
  copyA32B(lA + r * LPAD + hh * 16, src);
}

// B-tile: 64 N-rows x 32 halves from fp32 weight [N,K] with optional
// 128-block dequant scale. thread -> (n=tid>>2, q=tid&3 : 8 halves)
__device__ __forceinline__ void loadBtile(unsigned short* lB,
                                          const float* W, int ldb, int kmax,
                                          const float* S, int sldb,
                                          int nBase, int kb, int tid) {
  const int n = tid >> 2, q = tid & 3;
  const int f = nBase + n;
  const int k = kb + q * 8;
  const float* src = W + (size_t)f * ldb + k;
  const float sc = (S != nullptr) ? S[(f >> 7) * sldb + (k >> 7)] : 1.0f;
  f32x4 a = *(const f32x4*)(src);
  f32x4 b = *(const f32x4*)(src + 4);
  if (k + 32 < kmax) __builtin_prefetch(src + 32, 0, 1);  // next K-step cacheline
  H8 o;
  o.h[0] = f2bf(a[0] * sc); o.h[1] = f2bf(a[1] * sc);
  o.h[2] = f2bf(a[2] * sc); o.h[3] = f2bf(a[3] * sc);
  o.h[4] = f2bf(b[0] * sc); o.h[5] = f2bf(b[1] * sc);
  o.h[6] = f2bf(b[2] * sc); o.h[7] = f2bf(b[3] * sc);
  *(u32x4*)(lB + n * LPAD + q * 8) = o.q;
}

// --- per-lane fragment loads (ISA 7.12.2 layouts) --------------------------
__device__ __forceinline__ Frag16 loadAfrag(const unsigned short* lA, int w, int lane) {
  const unsigned short* ar = lA + (w * 16 + (lane & 15)) * LPAD;
  const int koff = (lane < 16) ? 0 : 8;      // lo lanes K{0..7,16..23}, hi K{8..15,24..31}
  Frag16 f;
  f.q[0] = *(const u32x4*)(ar + koff);
  f.q[1] = *(const u32x4*)(ar + koff + 16);
  return f;
}

__device__ __forceinline__ Frag16 loadBfrag(const unsigned short* lB, int n, int lane) {
  const unsigned short* br = lB + (n * 16 + (lane & 15)) * LPAD;
  const int koff = (lane < 16) ? 0 : 8;      // lo lanes K{0..7,16..23}, hi K{8..15,24..31}
  Frag16 f;
  f.q[0] = *(const u32x4*)(br + koff);
  f.q[1] = *(const u32x4*)(br + koff + 16);
  return f;
}

// ---------------------------------------------------------------------------
// small kernels
// ---------------------------------------------------------------------------
__global__ __launch_bounds__(256) void init_kernel(float* out, int* counts) {
  const int i = blockIdx.x * 256 + threadIdx.x;
  out[i] = 0.f;
  if (i < E_) counts[i] = 0;
}

__global__ __launch_bounds__(256) void tobf16_kernel(const float* __restrict__ x,
                                                     unsigned short* __restrict__ xb) {
  const int i = blockIdx.x * 256 + threadIdx.x;
  xb[i] = f2bf(x[i]);
}

// router: logits = x @ router_w.T ; softmax -> top8 -> renorm.
// also fuses the shared-expert sigmoid gate dot product.
__global__ __launch_bounds__(256)
void router_kernel(const float* __restrict__ x, const float* __restrict__ rweight,
                   const float* __restrict__ wshg,
                   float* __restrict__ rw, int* __restrict__ sel,
                   float* __restrict__ shg) {
  __shared__ float red[E_];
  __shared__ float red2;
  const int t = blockIdx.x;
  const int tid = threadIdx.x;
  const int e = tid & 31, chunk = tid >> 5;   // 8 chunks x 256 elems = D_
  if (tid < E_) red[tid] = 0.f;
  if (tid == 0) red2 = 0.f;
  __syncthreads();
  const float* xr = x + (size_t)t * D_ + chunk * 256;
  const float* wr = rweight + (size_t)e * D_ + chunk * 256;
  const float* wg = wshg + chunk * 256;
  float acc = 0.f, acc2 = 0.f;
  for (int i = 0; i < 256; ++i) {
    const float xv = xr[i];
    acc += xv * wr[i];
    if (e == 0) acc2 += xv * wg[i];
  }
  atomicAdd(&red[e], acc);
  if (e == 0) atomicAdd(&red2, acc2);
  __syncthreads();
  if (tid == 0) {
    float p[E_];
    float mx = red[0];
    for (int i = 1; i < E_; ++i) mx = fmaxf(mx, red[i]);
    for (int i = 0; i < E_; ++i) p[i] = __expf(red[i] - mx);
    unsigned int used = 0;
    float wsum = 0.f;
    int ids[TOPK_]; float vals[TOPK_];
    for (int kk = 0; kk < TOPK_; ++kk) {
      int bi = 0; float bv = -1.f;
      for (int i = 0; i < E_; ++i)
        if (!((used >> i) & 1u) && p[i] > bv) { bv = p[i]; bi = i; }
      used |= 1u << bi;
      ids[kk] = bi; vals[kk] = bv; wsum += bv;
    }
    const float inv = 1.f / wsum;   // one scalar divide per token
    for (int kk = 0; kk < TOPK_; ++kk) {
      sel[t * TOPK_ + kk] = ids[kk];
      rw[t * TOPK_ + kk]  = vals[kk] * inv;
    }
    shg[t] = fast_sigmoid(red2);
  }
}

__global__ __launch_bounds__(256)
void dispatch_kernel(const int* __restrict__ sel, int* __restrict__ counts,
                     int* __restrict__ slot_tok) {
  const int b = blockIdx.x * 256 + threadIdx.x;
  if (b >= T_ * TOPK_) return;
  const int e = sel[b];
  const int pos = atomicAdd(&counts[e], 1);
  if (pos < CAP_) slot_tok[e * CAP_ + pos] = b;
}

// ---------------------------------------------------------------------------
// WMMA GEMM kernels: block = 256 threads (8 waves), tile M=128 N=64 K=32
// ---------------------------------------------------------------------------
__global__ __launch_bounds__(256)
void shared_gateup_wmma(const unsigned short* __restrict__ xb,
                        const float* __restrict__ Wg,
                        const float* __restrict__ Wu,
                        unsigned short* __restrict__ H) {
  __shared__ __align__(16) unsigned short lA[128 * LPAD];
  __shared__ __align__(16) unsigned short lBg[64 * LPAD];
  __shared__ __align__(16) unsigned short lBu[64 * LPAD];
  const int tid = threadIdx.x;
  const int w = tid >> 5, lane = tid & 31;
  const int nBase = blockIdx.x * 64;
  const int mBase = blockIdx.y * 128;
  v8f accG[4], accU[4];
#pragma unroll
  for (int n = 0; n < 4; ++n) { accG[n] = vzero8(); accU[n] = vzero8(); }
  for (int kb = 0; kb < D_; kb += 32) {
    __syncthreads();
    loadAtileDirect(lA, xb, D_, mBase, kb, tid);
    loadBtile(lBg, Wg, D_, D_, nullptr, 0, nBase, kb, tid);
    loadBtile(lBu, Wu, D_, D_, nullptr, 0, nBase, kb, tid);
    ASYNC_WAIT();
    __syncthreads();
    Frag16 af = loadAfrag(lA, w, lane);
#pragma unroll
    for (int n = 0; n < 4; ++n) {
      Frag16 bg = loadBfrag(lBg, n, lane);
      accG[n] = wmma_bf16(af, bg, accG[n]);
      Frag16 bu = loadBfrag(lBu, n, lane);
      accU[n] = wmma_bf16(af, bu, accU[n]);
    }
  }
  const int mrow  = mBase + w * 16 + ((lane < 16) ? 0 : 8);
  const int ncol0 = nBase + (lane & 15);
#pragma unroll
  for (int n = 0; n < 4; ++n)
#pragma unroll
    for (int i = 0; i < 8; ++i) {
      const float g = accG[n][i], u = accU[n][i];
      const float hv = g * fast_sigmoid(g) * u;   // silu(g)*up
      H[(size_t)(mrow + i) * DSH_ + (ncol0 + n * 16)] = f2bf(hv);
    }
}

__global__ __launch_bounds__(256)
void shared_down_wmma(const unsigned short* __restrict__ Hsh,
                      const float* __restrict__ Wd,
                      const float* __restrict__ shg,
                      float* __restrict__ out) {
  __shared__ __align__(16) unsigned short lA[128 * LPAD];
  __shared__ __align__(16) unsigned short lB[64 * LPAD];
  const int tid = threadIdx.x;
  const int w = tid >> 5, lane = tid & 31;
  const int nBase = blockIdx.x * 64;
  const int mBase = blockIdx.y * 128;
  v8f acc[4];
#pragma unroll
  for (int n = 0; n < 4; ++n) acc[n] = vzero8();
  for (int kb = 0; kb < DSH_; kb += 32) {
    __syncthreads();
    loadAtileDirect(lA, Hsh, DSH_, mBase, kb, tid);
    loadBtile(lB, Wd, DSH_, DSH_, nullptr, 0, nBase, kb, tid);
    ASYNC_WAIT();
    __syncthreads();
    Frag16 af = loadAfrag(lA, w, lane);
#pragma unroll
    for (int n = 0; n < 4; ++n) {
      Frag16 bf = loadBfrag(lB, n, lane);
      acc[n] = wmma_bf16(af, bf, acc[n]);
    }
  }
  const int mrow  = mBase + w * 16 + ((lane < 16) ? 0 : 8);
  const int ncol0 = nBase + (lane & 15);
#pragma unroll
  for (int n = 0; n < 4; ++n)
#pragma unroll
    for (int i = 0; i < 8; ++i) {
      const int t = mrow + i;
      out[(size_t)t * D_ + (ncol0 + n * 16)] += shg[t] * acc[n][i];
    }
}

__global__ __launch_bounds__(256)
void expert_gateup_wmma(const unsigned short* __restrict__ xb,
                        const float* __restrict__ wg, const float* __restrict__ sg,
                        const float* __restrict__ wu, const float* __restrict__ su,
                        const int* __restrict__ slot_tok, const int* __restrict__ counts,
                        unsigned short* __restrict__ h_exp) {
  const int e = blockIdx.z;
  int count = counts[e]; if (count > CAP_) count = CAP_;
  const int mBase = blockIdx.y * 128;
  if (mBase >= count) return;
  __shared__ __align__(16) unsigned short lA[128 * LPAD];
  __shared__ __align__(16) unsigned short lBg[64 * LPAD];
  __shared__ __align__(16) unsigned short lBu[64 * LPAD];
  const int tid = threadIdx.x;
  const int w = tid >> 5, lane = tid & 31;
  const int nBase = blockIdx.x * 64;
  const float* Wg = wg + (size_t)e * DF_ * D_;
  const float* Wu = wu + (size_t)e * DF_ * D_;
  const float* Sg = sg + (size_t)e * (DF_ / 128) * (D_ / 128);
  const float* Su = su + (size_t)e * (DF_ / 128) * (D_ / 128);
  const int* slots = slot_tok + e * CAP_;
  unsigned short* H = h_exp + (size_t)e * CAP_ * DF_;
  v8f accG[4], accU[4];
#pragma unroll
  for (int n = 0; n < 4; ++n) { accG[n] = vzero8(); accU[n] = vzero8(); }
  for (int kb = 0; kb < D_; kb += 32) {
    __syncthreads();
    loadAtileGather(lA, xb, slots, count, mBase, kb, tid);
    loadBtile(lBg, Wg, D_, D_, Sg, D_ / 128, nBase, kb, tid);
    loadBtile(lBu, Wu, D_, D_, Su, D_ / 128, nBase, kb, tid);
    ASYNC_WAIT();
    __syncthreads();
    Frag16 af = loadAfrag(lA, w, lane);
#pragma unroll
    for (int n = 0; n < 4; ++n) {
      Frag16 bg = loadBfrag(lBg, n, lane);
      accG[n] = wmma_bf16(af, bg, accG[n]);
      Frag16 bu = loadBfrag(lBu, n, lane);
      accU[n] = wmma_bf16(af, bu, accU[n]);
    }
  }
  const int mrow  = mBase + w * 16 + ((lane < 16) ? 0 : 8);
  const int ncol0 = nBase + (lane & 15);
#pragma unroll
  for (int n = 0; n < 4; ++n)
#pragma unroll
    for (int i = 0; i < 8; ++i) {
      const float g = accG[n][i], u = accU[n][i];
      const float hv = g * fast_sigmoid(g) * u;
      H[(size_t)(mrow + i) * DF_ + (ncol0 + n * 16)] = f2bf(hv);
    }
}

__global__ __launch_bounds__(256)
void expert_down_wmma(const unsigned short* __restrict__ h_exp,
                      const float* __restrict__ wd, const float* __restrict__ sd,
                      const int* __restrict__ slot_tok, const int* __restrict__ counts,
                      const float* __restrict__ rw,
                      float* __restrict__ out) {
  const int e = blockIdx.z;
  int count = counts[e]; if (count > CAP_) count = CAP_;
  const int mBase = blockIdx.y * 128;
  if (mBase >= count) return;
  __shared__ __align__(16) unsigned short lA[128 * LPAD];
  __shared__ __align__(16) unsigned short lB[64 * LPAD];
  const int tid = threadIdx.x;
  const int w = tid >> 5, lane = tid & 31;
  const int nBase = blockIdx.x * 64;
  const unsigned short* A = h_exp + (size_t)e * CAP_ * DF_;
  const float* Wd = wd + (size_t)e * D_ * DF_;
  const float* Sd = sd + (size_t)e * (D_ / 128) * (DF_ / 128);
  const int* slots = slot_tok + e * CAP_;
  v8f acc[4];
#pragma unroll
  for (int n = 0; n < 4; ++n) acc[n] = vzero8();
  for (int kb = 0; kb < DF_; kb += 32) {
    __syncthreads();
    loadAtileDirect(lA, A, DF_, mBase, kb, tid);
    loadBtile(lB, Wd, DF_, DF_, Sd, DF_ / 128, nBase, kb, tid);
    ASYNC_WAIT();
    __syncthreads();
    Frag16 af = loadAfrag(lA, w, lane);
#pragma unroll
    for (int n = 0; n < 4; ++n) {
      Frag16 bf = loadBfrag(lB, n, lane);
      acc[n] = wmma_bf16(af, bf, acc[n]);
    }
  }
  const int slotBase = mBase + w * 16 + ((lane < 16) ? 0 : 8);
  const int ncol0    = nBase + (lane & 15);
#pragma unroll
  for (int n = 0; n < 4; ++n)
#pragma unroll
    for (int i = 0; i < 8; ++i) {
      const int slot = slotBase + i;
      if (slot < count) {
        const int b = slots[slot];
        const int t = b >> 3;                 // TOPK_==8
        atomicAdd(&out[(size_t)t * D_ + (ncol0 + n * 16)], rw[b] * acc[n][i]);
      }
    }
}

// ---------------------------------------------------------------------------
extern "C" void kernel_launch(void* const* d_in, const int* in_sizes, int n_in,
                              void* d_out, int out_size, void* d_ws, size_t ws_size,
                              hipStream_t stream) {
  (void)in_sizes; (void)n_in; (void)out_size; (void)ws_size;
  const float* x    = (const float*)d_in[0];   // [T, D]
  const float* rtw  = (const float*)d_in[1];   // [E, D]
  const float* wg   = (const float*)d_in[2];   // [E, DF, D]
  const float* sg   = (const float*)d_in[3];   // [E, DF/128, D/128]
  const float* wu   = (const float*)d_in[4];
  const float* su   = (const float*)d_in[5];
  const float* wd   = (const float*)d_in[6];   // [E, D, DF]
  const float* sd   = (const float*)d_in[7];   // [E, D/128, DF/128]
  const float* wsgt = (const float*)d_in[8];   // ws_gate [DSH, D]
  const float* wsup = (const float*)d_in[9];   // ws_up   [DSH, D]
  const float* wsdn = (const float*)d_in[10];  // ws_down [D, DSH]
  const float* wshg = (const float*)d_in[11];  // [1, D]
  float* out = (float*)d_out;

  char* ws = (char*)d_ws;
  size_t off = 0;
  auto take = [&](size_t bytes) -> char* {
    char* p = ws + off;
    off += (bytes + 255) & ~(size_t)255;
    return p;
  };
  unsigned short* xb       = (unsigned short*)take((size_t)T_ * D_ * 2);
  unsigned short* h_sh     = (unsigned short*)take((size_t)T_ * DSH_ * 2);
  unsigned short* h_exp    = (unsigned short*)take((size_t)E_ * CAP_ * DF_ * 2);
  float*          rwb      = (float*)take((size_t)T_ * TOPK_ * 4);
  int*            selb     = (int*)take((size_t)T_ * TOPK_ * 4);
  float*          shg      = (float*)take((size_t)T_ * 4);
  int*            counts   = (int*)take((size_t)E_ * 4);
  int*            slot_tok = (int*)take((size_t)E_ * CAP_ * 4);

  // 1. zero output + counters
  init_kernel<<<(T_ * D_) / 256, 256, 0, stream>>>(out, counts);
  // 2. router + shared sigmoid gate
  router_kernel<<<T_, 256, 0, stream>>>(x, rtw, wshg, rwb, selb, shg);
  // 3. capacity dispatch (slot -> flat (t,k))
  dispatch_kernel<<<(T_ * TOPK_ + 255) / 256, 256, 0, stream>>>(selb, counts, slot_tok);
  // 4. activations to bf16
  tobf16_kernel<<<(T_ * D_) / 256, 256, 0, stream>>>(x, xb);
  // 5. shared expert SwiGLU (gate/up fused)
  {
    dim3 g(DSH_ / 64, T_ / 128, 1);
    shared_gateup_wmma<<<g, 256, 0, stream>>>(xb, wsgt, wsup, h_sh);
  }
  // 6. shared expert down proj, gated add into out
  {
    dim3 g(D_ / 64, T_ / 128, 1);
    shared_down_wmma<<<g, 256, 0, stream>>>(h_sh, wsdn, shg, out);
  }
  // 7. routed experts gate/up fused (gathered A, block-scale dequant on B)
  {
    dim3 g(DF_ / 64, CAP_ / 128, E_);
    expert_gateup_wmma<<<g, 256, 0, stream>>>(xb, wg, sg, wu, su, slot_tok, counts, h_exp);
  }
  // 8. routed experts down proj + weighted scatter-add
  {
    dim3 g(D_ / 64, CAP_ / 128, E_);
    expert_down_wmma<<<g, 256, 0, stream>>>(h_exp, wd, sd, slot_tok, counts, rwb, out);
  }
}
